// VAE_recurrent_49495203119762
// MI455X (gfx1250) — matrix operand order repaired and checked by reading, not compiled, not run on hardware
//
#include <hip/hip_runtime.h>
#include <math.h>

// ---------------------------------------------------------------------------
// MI455X (gfx1250) recurrent-VAE forward pass.
//  - All GEMMs: v_wmma_f32_16x16x32_f16, operands pre-converted to f16 so the
//    K-loop staging is done entirely by the Tensor Data Mover
//    (tensor_load_to_lds, TENSORcnt-tracked, double-buffered LDS with
//    TDM-generated padding for conflict-free ds_read_b128).
//  - Tiny autoregressive chain: one thread per sample, weights in LDS.
// ---------------------------------------------------------------------------

typedef __attribute__((ext_vector_type(16))) _Float16     v16h;
typedef __attribute__((ext_vector_type(8)))  _Float16     v8h;
typedef __attribute__((ext_vector_type(8)))  float        v8f;
typedef __attribute__((ext_vector_type(4)))  unsigned int v4u;
typedef __attribute__((ext_vector_type(8)))  int          v8i_t;
typedef __attribute__((ext_vector_type(4)))  int          v4i_t;

#define BM 128
#define BN 64
#define BK 32
#define LDA 40   // 32 halfs + TDM pad (4 dwords per 16-dword row) -> conflict-free b128
#define LDB 40

union AFrag { v16h v; v8h h[2]; };

__device__ __forceinline__ v8h ld_v8h(const _Float16* p) {
    return *reinterpret_cast<const v8h*>(p);
}

// ACT: 0 = none, 1 = tanh, 2 = relu
template <int ACT>
__device__ __forceinline__ float act_apply(float v) {
    if (ACT == 1) return tanhf(v);
    if (ACT == 2) return fmaxf(v, 0.f);
    return v;
}

// ---------------------------------------------------------------------------
// TDM: 2D f16 tile (tile_k x tile_rows, row stride = row_stride elems) from
// global to LDS at lds_off, padding 4 dwords after every 16 dwords stored
// (so a 32-half row lands on a 40-half LDS stride). OOB rows/cols read zero
// via tensor_dim = remaining extents.
// D# packing per CDNA5 ISA.
// This toolchain exposes the 6-arg builtin:
//   (uint32x4 g0, int32x8 g1, int32x4 g2, int32x4 g3, int32x8, i32 cpol)
// ---------------------------------------------------------------------------
__device__ __forceinline__ void tdm_load_2d(unsigned lds_off, const _Float16* gsrc,
                                            int krem, int rows_rem,
                                            unsigned tile_k, unsigned tile_rows,
                                            unsigned row_stride) {
    const unsigned long long ga = (unsigned long long)(size_t)gsrc;
    const unsigned td0 = (unsigned)(krem    < 0 ? 0 : krem);
    const unsigned td1 = (unsigned)(rows_rem < 0 ? 0 : rows_rem);

    v4u g0;
    g0[0] = 1u;                                                // count=1, user mode
    g0[1] = lds_off;                                           // lds_addr (bytes)
    g0[2] = (unsigned)ga;                                      // global_addr[31:0]
    g0[3] = ((unsigned)(ga >> 32) & 0x01FFFFFFu) | (2u << 30); // global_addr[56:32] | type=2

    v8i_t g1;
    g1[0] = (int)((1u << 16)      // data_size = 2 bytes
                | (1u << 20)      // pad_enable
                | (3u << 22)      // pad_interval: 16 dwords
                | (3u << 25));    // pad_amount:   4 dwords
    g1[1] = (int)((td0 & 0xFFFFu) << 16);                      // tensor_dim0[15:0]
    g1[2] = (int)((td0 >> 16) | ((td1 & 0xFFFFu) << 16));      // dim0[31:16] | dim1[15:0]
    g1[3] = (int)((td1 >> 16) | (tile_k << 16));               // dim1[31:16] | tile_dim0
    g1[4] = (int)tile_rows;                                    // tile_dim1 (tile_dim2=0)
    g1[5] = (int)row_stride;                                   // tensor_dim0_stride[31:0]
    g1[6] = 0;                                                 // stride[47:32] | dim1_stride
    g1[7] = 0;

    const v4i_t z4 = {0, 0, 0, 0};
    const v8i_t z8 = {0, 0, 0, 0, 0, 0, 0, 0};
    __builtin_amdgcn_tensor_load_to_lds(g0, g1, z4, z4, z8, 0);
}

// ---------------------------------------------------------------------------
// Fused GEMM: C = act(A(MxK,f16) * B(KxN) + bias). B supplied pre-transposed
// f16 as Bt (N x K). Block 256 thr = 8 waves; tile 128x64; wave 32x32 (2x2
// WMMA). TDM double-buffered staging, one barrier per K-step.
// ---------------------------------------------------------------------------
template <int ACT, bool OUT_HALF>
__global__ __launch_bounds__(256)
void wmma_gemm_tdm(const _Float16* __restrict__ A, const _Float16* __restrict__ Bt,
                   const float* __restrict__ bias, void* __restrict__ Cv,
                   int M, int N, int K) {
    __shared__ _Float16 sA[2][BM * LDA];
    __shared__ _Float16 sB[2][BN * LDB];

    const int tid  = threadIdx.x;
    const int lane = tid & 31;
    const int wave = tid >> 5;
    const int wM   = (wave >> 1) * 32;
    const int wN   = (wave & 1) * 32;
    const int bM0  = blockIdx.y * BM;
    const int bN0  = blockIdx.x * BN;
    const int hs   = lane >> 4;
    const int lm   = lane & 15;
    const bool issuer = (wave == 0);

    v8f acc[2][2] = {};

    const _Float16* Abase = A  + (size_t)bM0 * (size_t)K;
    const _Float16* Bbase = Bt + (size_t)bN0 * (size_t)K;

    if (issuer) {
        tdm_load_2d((unsigned)(size_t)&sA[0][0], Abase, K, M - bM0, BK, BM, (unsigned)K);
        tdm_load_2d((unsigned)(size_t)&sB[0][0], Bbase, K, N - bN0, BK, BN, (unsigned)K);
    }

    int buf = 0;
    for (int k0 = 0; k0 < K; k0 += BK, buf ^= 1) {
        if (issuer) __builtin_amdgcn_s_wait_tensorcnt(0);
        __syncthreads();   // buf data visible; all waves done reading buf^1

        const int nk = k0 + BK;
        if (nk < K && issuer) {
            tdm_load_2d((unsigned)(size_t)&sA[buf ^ 1][0], Abase + nk, K - nk, M - bM0,
                        BK, BM, (unsigned)K);
            tdm_load_2d((unsigned)(size_t)&sB[buf ^ 1][0], Bbase + nk, K - nk, N - bN0,
                        BK, BN, (unsigned)K);
        }

        // fragments per the 16-bit WMMA VGPR layouts
        AFrag af[2];
#pragma unroll
        for (int i = 0; i < 2; ++i) {
            const _Float16* p = &sA[buf][(wM + i * 16 + lm) * LDA + hs * 8];
            af[i].h[0] = ld_v8h(p);        // K = hs*8 .. +7
            af[i].h[1] = ld_v8h(p + 16);   // K = 16 + hs*8 .. +7
        }
        AFrag bf[2];
#pragma unroll
        for (int j = 0; j < 2; ++j) {
            const _Float16* p = &sB[buf][(wN + j * 16 + lm) * LDB + hs * 8];
            bf[j].h[0] = ld_v8h(p);        // K = hs*8 .. +7
            bf[j].h[1] = ld_v8h(p + 16);   // K = 16 + hs*8 .. +7
        }
#pragma unroll
        for (int i = 0; i < 2; ++i)
#pragma unroll
            for (int j = 0; j < 2; ++j)
                acc[i][j] = __builtin_amdgcn_wmma_f32_16x16x32_f16(
                    false, af[i].v, false, bf[j].v, (short)0, acc[i][j], false, false);
    }

    // epilogue: bias + activation
#pragma unroll
    for (int i = 0; i < 2; ++i) {
#pragma unroll
        for (int j = 0; j < 2; ++j) {
            const int col = bN0 + wN + j * 16 + lm;
            if (col < N) {
                const float bv = bias[col];
#pragma unroll
                for (int r = 0; r < 8; ++r) {
                    const int row = bM0 + wM + i * 16 + r + hs * 8;
                    float v = act_apply<ACT>(acc[i][j][r] + bv);
                    if (OUT_HALF)
                        ((_Float16*)Cv)[(size_t)row * (size_t)N + col] = (_Float16)v;
                    else
                        ((float*)Cv)[(size_t)row * (size_t)N + col] = v;
                }
            }
        }
    }
}

// ---------------------------------------------------------------------------
// Pre-passes: f32 -> f16 convert (for x) and f32 (KxN) -> f16 (NxK) transpose
// (for weights) so TDM can stage GEMM tiles without conversion.
// ---------------------------------------------------------------------------
__global__ __launch_bounds__(256)
void convert_f16_kernel(const float* __restrict__ src, _Float16* __restrict__ dst,
                        long long n) {
    const long long i0 = ((long long)blockIdx.x * 256 + threadIdx.x) * 8;
    if (i0 + 8 <= n) {
        const float4 f0 = reinterpret_cast<const float4*>(src + i0)[0];
        const float4 f1 = reinterpret_cast<const float4*>(src + i0)[1];
        v8h h;
        h[0] = (_Float16)f0.x; h[1] = (_Float16)f0.y; h[2] = (_Float16)f0.z; h[3] = (_Float16)f0.w;
        h[4] = (_Float16)f1.x; h[5] = (_Float16)f1.y; h[6] = (_Float16)f1.z; h[7] = (_Float16)f1.w;
        *reinterpret_cast<v8h*>(dst + i0) = h;
    } else {
        for (long long i = i0; i < n; ++i) dst[i] = (_Float16)src[i];
    }
}

__global__ __launch_bounds__(256)
void transpose_f16_kernel(const float* __restrict__ W, _Float16* __restrict__ Wt,
                          int K, int N) {
    __shared__ float tile[32][33];
    const int kb = blockIdx.y * 32, nb = blockIdx.x * 32;
    const int tx = threadIdx.x & 31, ty = threadIdx.x >> 5;  // 32 x 8
#pragma unroll
    for (int r = ty; r < 32; r += 8) {
        const int k = kb + r, n = nb + tx;
        tile[r][tx] = (k < K && n < N) ? W[(size_t)k * N + n] : 0.f;
    }
    __syncthreads();
#pragma unroll
    for (int r = ty; r < 32; r += 8) {
        const int n = nb + r, k = kb + tx;
        if (n < N && k < K)
            Wt[(size_t)n * K + k] = (_Float16)tile[tx][r];
    }
}

// ---------------------------------------------------------------------------
// Autoregressive chain: one thread per sample, tiny-MLP weights in LDS.
// Fuses var = exp(clip(logvar)), z = mu + var*eps, zt = tanh(z) (f16), and
// the u_rand passthrough output.
// ---------------------------------------------------------------------------
__global__ __launch_bounds__(256)
void chain_kernel(const float* __restrict__ u_rand, const float* __restrict__ mu,
                  const float* __restrict__ lv,
                  const float* __restrict__ gFinvW1, const float* __restrict__ gFinvb1,
                  const float* __restrict__ gFinvW2, const float* __restrict__ gFinvb2,
                  const float* __restrict__ gFW1, const float* __restrict__ gFb1,
                  const float* __restrict__ gFW2, const float* __restrict__ gFb2,
                  float* __restrict__ out_urand, float* __restrict__ out_eps,
                  float* __restrict__ out_uhat, float* __restrict__ out_var,
                  _Float16* __restrict__ zt, int Bn) {
    __shared__ float sFinvW1[4096];
    __shared__ float sFinvb1[256];
    __shared__ float sFinvW2[256];
    __shared__ float sFinvb2[16];
    __shared__ float sFW1[4096];
    __shared__ float sFb1[256];
    __shared__ float sFW2[256];
    __shared__ float sFb2[16];

    const int tid = threadIdx.x;
    for (int i = tid; i < 4096; i += 256) { sFinvW1[i] = gFinvW1[i]; sFW1[i] = gFW1[i]; }
    for (int i = tid; i < 256;  i += 256) { sFinvb1[i] = gFinvb1[i]; sFinvW2[i] = gFinvW2[i];
                                            sFb1[i] = gFb1[i];       sFW2[i] = gFW2[i]; }
    if (tid < 16) { sFinvb2[tid] = gFinvb2[tid]; sFb2[tid] = gFb2[tid]; }
    __syncthreads();

    const int b = blockIdx.x * 256 + tid;
    if (b >= Bn) return;

    float eps[16], uh[16], hb[16];

    for (int i = 0; i < 16; ++i) {
        const float u = u_rand[(size_t)b * 16 + i];
#pragma unroll
        for (int c = 0; c < 16; ++c) {
            float a = sFinvb1[i * 16 + c] + u * sFinvW1[(i * 16 + 0) * 16 + c];
            for (int j = 0; j < i; ++j)
                a += eps[j] * sFinvW1[(i * 16 + 1 + j) * 16 + c];
            hb[c] = 1.f / (1.f + __expf(-a));
        }
        float xi = sFinvb2[i];
#pragma unroll
        for (int c = 0; c < 16; ++c) xi += hb[c] * sFinvW2[i * 16 + c];
        for (int j = i; j > 0; --j) eps[j] = eps[j - 1];
        eps[0] = xi;
#pragma unroll
        for (int c = 0; c < 16; ++c) {
            float a = sFb1[i * 16 + c];
            for (int j = 0; j <= i; ++j)
                a += eps[j] * sFW1[(i * 16 + j) * 16 + c];
            hb[c] = 1.f / (1.f + __expf(-a));
        }
        float d = sFb2[i];
#pragma unroll
        for (int c = 0; c < 16; ++c) d += hb[c] * sFW2[i * 16 + c];
        uh[i] = d;
    }

#pragma unroll
    for (int j = 0; j < 16; ++j) {
        const size_t o = (size_t)b * 16 + j;
        const float m = mu[o];
        float l = lv[o];
        l = fminf(4.f, fmaxf(-4.f, l));
        const float v = __expf(l);
        const float z = m + v * eps[j];
        out_eps[o]   = eps[j];
        out_var[o]   = v;
        out_uhat[o]  = uh[j];
        out_urand[o] = u_rand[o];
        zt[o] = (_Float16)tanhf(z);
    }
}

// ---------------------------------------------------------------------------
extern "C" void kernel_launch(void* const* d_in, const int* in_sizes, int n_in,
                              void* d_out, int out_size, void* d_ws, size_t ws_size,
                              hipStream_t stream) {
    const float* x      = (const float*)d_in[0];
    const float* u_rand = (const float*)d_in[1];
    const float* W1   = (const float*)d_in[2];  const float* b1   = (const float*)d_in[3];
    const float* W2   = (const float*)d_in[4];  const float* b2   = (const float*)d_in[5];
    const float* Wmu1 = (const float*)d_in[6];  const float* bmu1 = (const float*)d_in[7];
    const float* Wmu2 = (const float*)d_in[8];  const float* bmu2 = (const float*)d_in[9];
    const float* Wlv1 = (const float*)d_in[10]; const float* blv1 = (const float*)d_in[11];
    const float* Wlv2 = (const float*)d_in[12]; const float* blv2 = (const float*)d_in[13];
    const float* Wd1  = (const float*)d_in[14]; const float* bd1  = (const float*)d_in[15];
    const float* Wd2  = (const float*)d_in[16]; const float* bd2  = (const float*)d_in[17];
    const float* Wd3  = (const float*)d_in[18]; const float* bd3  = (const float*)d_in[19];
    const float* Wd4  = (const float*)d_in[20]; const float* bd4  = (const float*)d_in[21];
    const float* FinvW1 = (const float*)d_in[22]; const float* Finvb1 = (const float*)d_in[23];
    const float* FinvW2 = (const float*)d_in[24]; const float* Finvb2 = (const float*)d_in[25];
    const float* FW1    = (const float*)d_in[26]; const float* Fb1    = (const float*)d_in[27];
    const float* FW2    = (const float*)d_in[28]; const float* Fb2    = (const float*)d_in[29];

    const int Bsz = 8192;

    // workspace layout (bytes)
    char* ws = (char*)d_ws;
    _Float16* xh    = (_Float16*)(ws + 0);          // 8192x3072 f16
    _Float16* W1t   = (_Float16*)(ws + 50331648);   // 512x3072
    _Float16* W2t   = (_Float16*)(ws + 53477376);   // 256x512
    _Float16* Wmu1t = (_Float16*)(ws + 53739520);   // 128x256
    _Float16* Wmu2t = (_Float16*)(ws + 53805056);   // 16x128
    _Float16* Wlv1t = (_Float16*)(ws + 53809152);   // 128x256
    _Float16* Wlv2t = (_Float16*)(ws + 53874688);   // 16x128
    _Float16* Wd1t  = (_Float16*)(ws + 53878784);   // 128x16
    _Float16* Wd2t  = (_Float16*)(ws + 53882880);   // 256x128
    _Float16* Wd3t  = (_Float16*)(ws + 53948416);   // 512x256
    _Float16* Wd4t  = (_Float16*)(ws + 54210560);   // 3072x512
    _Float16* c1    = (_Float16*)(ws + 57356288);   // 8192x512
    _Float16* h     = (_Float16*)(ws + 65744896);   // 8192x256
    _Float16* tmu   = (_Float16*)(ws + 69939200);   // 8192x128
    _Float16* tlv   = (_Float16*)(ws + 72036352);   // 8192x128
    float*    mu    = (float*)   (ws + 74133504);   // 8192x16
    float*    lvr   = (float*)   (ws + 74657792);   // 8192x16
    _Float16* zt    = (_Float16*)(ws + 75182080);   // 8192x16
    _Float16* a1    = (_Float16*)(ws + 75444224);   // 8192x128
    _Float16* a2    = (_Float16*)(ws + 77541376);   // 8192x256
    _Float16* a3    = (_Float16*)(ws + 81735680);   // 8192x512

    // output layout: (x_rec, u_rand, eps, u_hat, var) flattened
    float* out     = (float*)d_out;
    float* x_rec   = out;
    float* o_urand = out + (size_t)Bsz * 3072;
    float* o_eps   = o_urand + (size_t)Bsz * 16;
    float* o_uhat  = o_eps   + (size_t)Bsz * 16;
    float* o_var   = o_uhat  + (size_t)Bsz * 16;

    dim3 blk(256);
    auto grid_for = [](int M, int N) { return dim3((unsigned)((N + BN - 1) / BN),
                                                   (unsigned)((M + BM - 1) / BM)); };
    auto tgrid = [](int K, int N) { return dim3((unsigned)((N + 31) / 32),
                                                (unsigned)((K + 31) / 32)); };

    // pre-passes: f16 copies / transposed f16 weights for TDM staging
    const long long nx = (long long)Bsz * 3072;
    convert_f16_kernel<<<dim3((unsigned)((nx + 2047) / 2048)), blk, 0, stream>>>(x, xh, nx);
    transpose_f16_kernel<<<tgrid(3072, 512), blk, 0, stream>>>(W1,   W1t,   3072, 512);
    transpose_f16_kernel<<<tgrid(512, 256),  blk, 0, stream>>>(W2,   W2t,   512,  256);
    transpose_f16_kernel<<<tgrid(256, 128),  blk, 0, stream>>>(Wmu1, Wmu1t, 256,  128);
    transpose_f16_kernel<<<tgrid(128, 16),   blk, 0, stream>>>(Wmu2, Wmu2t, 128,  16);
    transpose_f16_kernel<<<tgrid(256, 128),  blk, 0, stream>>>(Wlv1, Wlv1t, 256,  128);
    transpose_f16_kernel<<<tgrid(128, 16),   blk, 0, stream>>>(Wlv2, Wlv2t, 128,  16);
    transpose_f16_kernel<<<tgrid(16, 128),   blk, 0, stream>>>(Wd1,  Wd1t,  16,   128);
    transpose_f16_kernel<<<tgrid(128, 256),  blk, 0, stream>>>(Wd2,  Wd2t,  128,  256);
    transpose_f16_kernel<<<tgrid(256, 512),  blk, 0, stream>>>(Wd3,  Wd3t,  256,  512);
    transpose_f16_kernel<<<tgrid(512, 3072), blk, 0, stream>>>(Wd4,  Wd4t,  512,  3072);

    // encoder
    wmma_gemm_tdm<1, true><<<grid_for(Bsz, 512),  blk, 0, stream>>>(xh,  W1t,   b1,   c1,  Bsz, 512, 3072);
    wmma_gemm_tdm<2, true><<<grid_for(Bsz, 256),  blk, 0, stream>>>(c1,  W2t,   b2,   h,   Bsz, 256, 512);
    wmma_gemm_tdm<1, true><<<grid_for(Bsz, 128),  blk, 0, stream>>>(h,   Wmu1t, bmu1, tmu, Bsz, 128, 256);
    wmma_gemm_tdm<0, false><<<grid_for(Bsz, 16),  blk, 0, stream>>>(tmu, Wmu2t, bmu2, mu,  Bsz, 16,  128);
    wmma_gemm_tdm<1, true><<<grid_for(Bsz, 128),  blk, 0, stream>>>(h,   Wlv1t, blv1, tlv, Bsz, 128, 256);
    wmma_gemm_tdm<0, false><<<grid_for(Bsz, 16),  blk, 0, stream>>>(tlv, Wlv2t, blv2, lvr, Bsz, 16,  128);

    // autoregressive chain + reparameterization + tanh(z)
    chain_kernel<<<dim3(Bsz / 256), blk, 0, stream>>>(
        u_rand, mu, lvr, FinvW1, Finvb1, FinvW2, Finvb2, FW1, Fb1, FW2, Fb2,
        o_urand, o_eps, o_uhat, o_var, zt, Bsz);

    // decoder
    wmma_gemm_tdm<1, true><<<grid_for(Bsz, 128),   blk, 0, stream>>>(zt, Wd1t, bd1, a1,    Bsz, 128,  16);
    wmma_gemm_tdm<1, true><<<grid_for(Bsz, 256),   blk, 0, stream>>>(a1, Wd2t, bd2, a2,    Bsz, 256,  128);
    wmma_gemm_tdm<1, true><<<grid_for(Bsz, 512),   blk, 0, stream>>>(a2, Wd3t, bd3, a3,    Bsz, 512,  256);
    wmma_gemm_tdm<0, false><<<grid_for(Bsz, 3072), blk, 0, stream>>>(a3, Wd4t, bd4, x_rec, Bsz, 3072, 512);
}
